// Model_22728966930624
// MI455X (gfx1250) — matrix operand order repaired and checked by reading, not compiled, not run on hardware
//
#include <hip/hip_runtime.h>

typedef _Float16 f16;
typedef __attribute__((ext_vector_type(16))) _Float16 v16h;
typedef __attribute__((ext_vector_type(8)))  _Float16 v8h;
typedef __attribute__((ext_vector_type(8)))  float    v8f;

#define S_LEN   4096
#define DMODEL  768
#define NHEAD   12
#define DHEAD   64
#define NLAYER  4
#define FFDIM   3072
#define CHUNK   256
#define NCHUNK  16

__device__ __forceinline__ v8f wmma16(v16h a, v16h b, v8f c) {
    // D = A(16x32 f16) * B(32x16 f16) + C(16x16 f32)
    return __builtin_amdgcn_wmma_f32_16x16x32_f16(false, a, false, b, (short)0, c, false, false);
}

__device__ __forceinline__ v8f zero8() {
    v8f z;
#pragma unroll
    for (int i = 0; i < 8; ++i) z[i] = 0.0f;
    return z;
}

__device__ __forceinline__ v16h ld16(const f16* p0, const f16* p1) {
    v8h a = *(const v8h*)p0;
    v8h b = *(const v8h*)p1;
    return __builtin_shufflevector(a, b, 0,1,2,3,4,5,6,7,8,9,10,11,12,13,14,15);
}

// LDS byte offset of a __shared__ pointer (flat aperture low 32 bits == LDS offset).
__device__ __forceinline__ unsigned lds_off(const void* p) {
    return (unsigned)(unsigned long long)p;
}

// Async global->LDS 16-byte copy, tracked with ASYNCcnt (CDNA5).
__device__ __forceinline__ void async_b128(unsigned lds, const void* g) {
    asm volatile("global_load_async_to_lds_b128 %0, %1, off"
                 :: "v"(lds), "v"(g) : "memory");
}
__device__ __forceinline__ void wait_async0() {
    asm volatile("s_wait_asynccnt 0x0" ::: "memory");
}

// ---------------------------------------------------------------------------
// Weight transpose + f16 convert: W[K,N] fp32 -> Wt[N,K] f16
// ---------------------------------------------------------------------------
__global__ __launch_bounds__(256)
void wtrans_kernel(const float* __restrict__ W, f16* __restrict__ Wt, int K, int N) {
    __shared__ float tile[32][33];
    int n0 = blockIdx.x * 32, k0 = blockIdx.y * 32;
    int t = threadIdx.x;
#pragma unroll
    for (int i = 0; i < 4; ++i) {
        int idx = t + i * 256;
        int kk = idx >> 5, nn = idx & 31;
        tile[kk][nn] = W[(size_t)(k0 + kk) * N + n0 + nn];
    }
    __syncthreads();
#pragma unroll
    for (int i = 0; i < 4; ++i) {
        int idx = t + i * 256;
        int nn = idx >> 5, kk = idx & 31;
        Wt[(size_t)(n0 + nn) * K + k0 + kk] = (f16)tile[kk][nn];
    }
}

// ---------------------------------------------------------------------------
// GEMM: out = act((A[M,K](f16) @ Wt[N,K](f16,transposed) + bias) * scale)
// 256 threads = 8 waves (4 row x 2 col); wave computes 32x64 (2x4 WMMA tiles).
// Double-buffered LDS tiles filled by async global->LDS copies.
// ---------------------------------------------------------------------------
__global__ __launch_bounds__(256)
void gemm_kernel(const f16* __restrict__ A, const f16* __restrict__ Wt,
                 const float* __restrict__ bias,
                 float* __restrict__ outf, f16* __restrict__ outh,
                 int M, int N, int K, float scale, int act /*0 none, 1 gelu*/) {
    __shared__ __align__(16) f16 As[2][128 * 40];  // [m][k] stride 40
    __shared__ __align__(16) f16 Bs[2][128 * 40];  // [n][k] stride 40
    const int t    = threadIdx.x;
    const int lane = t & 31;
    const int l15  = lane & 15;
    const int hi   = lane >> 4;
    const int wid  = t >> 5;
    const int wr   = wid & 3;
    const int wc   = wid >> 2;
    const int m0   = blockIdx.y * 128;
    const int n0   = blockIdx.x * 128;
    const int cr   = t >> 2, cc = (t & 3) * 8;   // copy row / chunk-col (halves)
    const int nk   = K >> 5;

    v8f acc[2][4];
#pragma unroll
    for (int i = 0; i < 2; ++i)
#pragma unroll
        for (int j = 0; j < 4; ++j) acc[i][j] = zero8();

    // issue copies for tile 0
#pragma unroll
    for (int half = 0; half < 2; ++half) {
        int r = cr + half * 64;
        async_b128(lds_off(&As[0][r * 40 + cc]), A + (size_t)(m0 + r) * K + cc);
        async_b128(lds_off(&Bs[0][r * 40 + cc]), Wt + (size_t)(n0 + r) * K + cc);
    }

    for (int kt = 0; kt < nk; ++kt) {
        const int cur = kt & 1, nxt = cur ^ 1;
        wait_async0();
        __syncthreads();
        if (kt + 1 < nk) {
            int k0 = (kt + 1) << 5;
#pragma unroll
            for (int half = 0; half < 2; ++half) {
                int r = cr + half * 64;
                async_b128(lds_off(&As[nxt][r * 40 + cc]), A + (size_t)(m0 + r) * K + k0 + cc);
                async_b128(lds_off(&Bs[nxt][r * 40 + cc]), Wt + (size_t)(n0 + r) * K + k0 + cc);
            }
        }
        v16h af[2], bf[4];
#pragma unroll
        for (int i = 0; i < 2; ++i) {
            const f16* p = &As[cur][(wr * 32 + i * 16 + l15) * 40];
            af[i] = ld16(p + 16 * hi, p + 16 * hi + 8);
        }
#pragma unroll
        for (int j = 0; j < 4; ++j) {
            const f16* p = &Bs[cur][(wc * 64 + j * 16 + l15) * 40 + 16 * hi];
            bf[j] = ld16(p, p + 8);
        }
#pragma unroll
        for (int i = 0; i < 2; ++i)
#pragma unroll
            for (int j = 0; j < 4; ++j) acc[i][j] = wmma16(af[i], bf[j], acc[i][j]);
    }

#pragma unroll
    for (int i = 0; i < 2; ++i)
#pragma unroll
        for (int j = 0; j < 4; ++j) {
            int nn = n0 + wc * 64 + j * 16 + l15;
            float bv = bias ? bias[nn] : 0.0f;
#pragma unroll
            for (int e = 0; e < 8; ++e) {
                int mm = m0 + wr * 32 + i * 16 + e + 8 * hi;
                float y = (acc[i][j][e] + bv) * scale;
                if (act == 1) y = 0.5f * y * (1.0f + erff(y * 0.70710678118f));
                if (outf) outf[(size_t)mm * N + nn] = y;
                if (outh) outh[(size_t)mm * N + nn] = (f16)y;
            }
        }
}

// ---------------------------------------------------------------------------
// Banded flash attention: one block per (chunk n, head h); f16 in / f16 out.
// ---------------------------------------------------------------------------
__global__ __launch_bounds__(256)
void attn_kernel(const f16* __restrict__ q, const f16* __restrict__ kbuf,
                 const f16* __restrict__ vbuf, const int* __restrict__ amask,
                 f16* __restrict__ out) {
    __shared__ __align__(16) f16 Ks[32 * 72];       // [key][d]
    __shared__ __align__(16) f16 Vt[64 * 40];       // [d][key]
    __shared__ __align__(16) f16 Pb[8 * 32 * 40];   // per-wave P [q][key]
    __shared__ int mb[32];

    const int n    = blockIdx.x;
    const int h    = blockIdx.y;
    const int t    = threadIdx.x;
    const int lane = t & 31;
    const int l15  = lane & 15;
    const int hi   = lane >> 4;
    const int wid  = t >> 5;
    const int qw   = wid * 32;

    // Preload Q fragments (A layout): 2 row tiles x 2 k-steps of DH=64.
    v16h qf[2][2];
#pragma unroll
    for (int tm = 0; tm < 2; ++tm) {
        int row = n * CHUNK + qw + tm * 16 + l15;
        const f16* qp = q + (size_t)row * DMODEL + h * DHEAD;
#pragma unroll
        for (int ds = 0; ds < 2; ++ds)
            qf[tm][ds] = ld16(qp + ds * 32 + 16 * hi, qp + ds * 32 + 16 * hi + 8);
    }

    v8f Oa[2][4];
    float mrun[2][8], lsum[2][8];
#pragma unroll
    for (int tm = 0; tm < 2; ++tm) {
#pragma unroll
        for (int j = 0; j < 4; ++j) Oa[tm][j] = zero8();
#pragma unroll
        for (int e = 0; e < 8; ++e) { mrun[tm][e] = -1e30f; lsum[tm][e] = 0.0f; }
    }

    f16* Pw = &Pb[wid * 32 * 40];
    const int key = t >> 3, cc = (t & 7) * 8;   // staging coords: 32 keys x 8 chunks

    for (int kt = 0; kt < 24; ++kt) {
        __syncthreads();
        int kg = n * CHUNK - CHUNK + kt * 32 + key;
        int kgc = kg < 0 ? 0 : (kg >= S_LEN ? S_LEN - 1 : kg);
        // async K tile copy (clamped addresses; OOB keys masked later)
        async_b128(lds_off(&Ks[key * 72 + cc]),
                   kbuf + (size_t)kgc * DMODEL + h * DHEAD + cc);
        // manual transposed V staging (overlaps the async copy)
        {
            v8h vv = *(const v8h*)(vbuf + (size_t)kgc * DMODEL + h * DHEAD + cc);
#pragma unroll
            for (int j = 0; j < 8; ++j) Vt[(cc + j) * 40 + key] = vv[j];
        }
        if (t < 32) {
            int kg2 = n * CHUNK - CHUNK + kt * 32 + t;
            mb[t] = (kg2 >= 0 && kg2 < S_LEN) ? amask[kg2] : 0;
        }
        wait_async0();
        __syncthreads();

        // scores: S[32q,32k] = Q @ K^T
        v8f sc[2][2];
#pragma unroll
        for (int tm = 0; tm < 2; ++tm)
#pragma unroll
            for (int tn = 0; tn < 2; ++tn) sc[tm][tn] = zero8();
#pragma unroll
        for (int ds = 0; ds < 2; ++ds) {
            v16h bf[2];
#pragma unroll
            for (int tn = 0; tn < 2; ++tn) {
                const f16* p = &Ks[(tn * 16 + l15) * 72 + ds * 32 + 16 * hi];
                bf[tn] = ld16(p, p + 8);
            }
#pragma unroll
            for (int tm = 0; tm < 2; ++tm)
#pragma unroll
                for (int tn = 0; tn < 2; ++tn)
                    sc[tm][tn] = wmma16(qf[tm][ds], bf[tn], sc[tm][tn]);
        }

        // band mask + online softmax; write P to per-wave LDS
#pragma unroll
        for (int tm = 0; tm < 2; ++tm) {
#pragma unroll
            for (int e = 0; e < 8; ++e) {
                int qi  = qw + tm * 16 + e + 8 * hi;
                int kj0 = kt * 32 + l15;
                int kj1 = kj0 + 16;
                bool v0 = (kj0 >= qi) && (kj0 <= qi + 512) && (mb[l15] != 0);
                bool v1 = (kj1 >= qi) && (kj1 <= qi + 512) && (mb[16 + l15] != 0);
                float s0 = v0 ? sc[tm][0][e] : -1e9f;
                float s1 = v1 ? sc[tm][1][e] : -1e9f;
                float rm = fmaxf(s0, s1);
#pragma unroll
                for (int off = 8; off > 0; off >>= 1)
                    rm = fmaxf(rm, __shfl_xor(rm, off, 32));
                float mold = mrun[tm][e];
                float mnew = fmaxf(mold, rm);
                float corr = __expf(mold - mnew);
                float p0 = v0 ? __expf(s0 - mnew) : 0.0f;
                float p1 = v1 ? __expf(s1 - mnew) : 0.0f;
                float rs = p0 + p1;
#pragma unroll
                for (int off = 8; off > 0; off >>= 1)
                    rs += __shfl_xor(rs, off, 32);
                lsum[tm][e] = lsum[tm][e] * corr + rs;
                mrun[tm][e] = mnew;
#pragma unroll
                for (int j = 0; j < 4; ++j) Oa[tm][j][e] *= corr;
                Pw[(tm * 16 + e + 8 * hi) * 40 + l15]      = (f16)p0;
                Pw[(tm * 16 + e + 8 * hi) * 40 + 16 + l15] = (f16)p1;
            }
        }
        __syncthreads();

        // O += P[32q,32k] @ V[32k,64d]
        v16h vf[4], pf[2];
#pragma unroll
        for (int jd = 0; jd < 4; ++jd) {
            const f16* p = &Vt[(jd * 16 + l15) * 40 + 16 * hi];
            vf[jd] = ld16(p, p + 8);
        }
#pragma unroll
        for (int tm = 0; tm < 2; ++tm) {
            const f16* p = &Pw[(tm * 16 + l15) * 40];
            pf[tm] = ld16(p + 16 * hi, p + 16 * hi + 8);
        }
#pragma unroll
        for (int tm = 0; tm < 2; ++tm)
#pragma unroll
            for (int jd = 0; jd < 4; ++jd)
                Oa[tm][jd] = wmma16(pf[tm], vf[jd], Oa[tm][jd]);
    }

#pragma unroll
    for (int tm = 0; tm < 2; ++tm)
#pragma unroll
        for (int e = 0; e < 8; ++e) {
            float inv = 1.0f / lsum[tm][e];
            int row = n * CHUNK + qw + tm * 16 + e + 8 * hi;
#pragma unroll
            for (int jd = 0; jd < 4; ++jd) {
                int col = h * DHEAD + jd * 16 + l15;
                out[(size_t)row * DMODEL + col] = (f16)(Oa[tm][jd][e] * inv);
            }
        }
}

// ---------------------------------------------------------------------------
// Elementwise / reduction kernels
// ---------------------------------------------------------------------------
__global__ void embed_kernel(const int* __restrict__ ids, const float* __restrict__ we,
                             const float* __restrict__ pe, float* __restrict__ out) {
    int i = blockIdx.x * 256 + threadIdx.x;
    int s = i / DMODEL;
    int d = i - s * DMODEL;
    out[i] = we[(size_t)ids[s] * DMODEL + d] + pe[i];
}

__global__ __launch_bounds__(256)
void ln_kernel(const float* __restrict__ x, const float* __restrict__ res,
               const float* __restrict__ g, const float* __restrict__ b,
               float* __restrict__ out, f16* __restrict__ outh) {
    __shared__ float red[256];
    int row = blockIdx.x, t = threadIdx.x;
    const float* xr = x + (size_t)row * DMODEL;
    float v0 = xr[t], v1 = xr[t + 256], v2 = xr[t + 512];
    if (res) {
        const float* rr = res + (size_t)row * DMODEL;
        v0 += rr[t]; v1 += rr[t + 256]; v2 += rr[t + 512];
    }
    red[t] = v0 + v1 + v2;
    __syncthreads();
    for (int off = 128; off > 0; off >>= 1) {
        if (t < off) red[t] += red[t + off];
        __syncthreads();
    }
    float mean = red[0] * (1.0f / 768.0f);
    __syncthreads();
    float d0 = v0 - mean, d1 = v1 - mean, d2 = v2 - mean;
    red[t] = d0 * d0 + d1 * d1 + d2 * d2;
    __syncthreads();
    for (int off = 128; off > 0; off >>= 1) {
        if (t < off) red[t] += red[t + off];
        __syncthreads();
    }
    float rstd = rsqrtf(red[0] * (1.0f / 768.0f) + 1e-5f);
    float y0 = d0 * rstd * g[t]       + b[t];
    float y1 = d1 * rstd * g[t + 256] + b[t + 256];
    float y2 = d2 * rstd * g[t + 512] + b[t + 512];
    float* o = out + (size_t)row * DMODEL;
    o[t] = y0; o[t + 256] = y1; o[t + 512] = y2;
    if (outh) {
        f16* oh = outh + (size_t)row * DMODEL;
        oh[t] = (f16)y0; oh[t + 256] = (f16)y1; oh[t + 512] = (f16)y2;
    }
}

__global__ __launch_bounds__(256)
void pool_kernel(const float* __restrict__ x, const int* __restrict__ amask,
                 float* __restrict__ pooled) {
    __shared__ float rx[256];
    __shared__ float rm[256];
    int d = blockIdx.x, t = threadIdx.x;
    float sx = 0.0f, sm = 0.0f;
    for (int s = t; s < S_LEN; s += 256) {
        float mf = (float)amask[s];
        sx += x[(size_t)s * DMODEL + d] * mf;
        sm += mf;
    }
    rx[t] = sx; rm[t] = sm;
    __syncthreads();
    for (int off = 128; off > 0; off >>= 1) {
        if (t < off) { rx[t] += rx[t + off]; rm[t] += rm[t + off]; }
        __syncthreads();
    }
    if (t == 0) pooled[d] = rx[0] / fmaxf(rm[0], 1e-9f);
}

__global__ void fc_kernel(const float* __restrict__ in, const float* __restrict__ W,
                          const float* __restrict__ bias, float* __restrict__ out,
                          int K, int N, int relu) {
    int nn = blockIdx.x * blockDim.x + threadIdx.x;
    if (nn >= N) return;
    float a = bias[nn];
    for (int k = 0; k < K; ++k) a += in[k] * W[(size_t)k * N + nn];
    out[nn] = relu ? fmaxf(a, 0.0f) : a;
}

__global__ void final_kernel(const float* __restrict__ h2, const float* __restrict__ W3,
                             const float* __restrict__ b3, float* __restrict__ out) {
    if (threadIdx.x == 0) {
        float a = b3[0];
        for (int k = 0; k < 250; ++k) a += h2[k] * W3[k];
        out[0] = a;
        out[1] = a;
    }
}

// ---------------------------------------------------------------------------
extern "C" void kernel_launch(void* const* d_in, const int* in_sizes, int n_in,
                              void* d_out, int out_size, void* d_ws, size_t ws_size,
                              hipStream_t stream) {
    (void)in_sizes; (void)n_in; (void)out_size; (void)ws_size;
    const int*   ids     = (const int*)d_in[0];
    const int*   amask   = (const int*)d_in[1];
    const float* wordemb = (const float*)d_in[2];
    const float* posemb  = (const float*)d_in[3];
    const float* emb_g   = (const float*)d_in[4];
    const float* emb_b   = (const float*)d_in[5];
    const float* Wq = (const float*)d_in[6];
    const float* Wk = (const float*)d_in[7];
    const float* Wv = (const float*)d_in[8];
    const float* Wo = (const float*)d_in[9];
    const float* bq = (const float*)d_in[10];
    const float* bk = (const float*)d_in[11];
    const float* bv = (const float*)d_in[12];
    const float* bo = (const float*)d_in[13];
    const float* ln1g = (const float*)d_in[14];
    const float* ln1b = (const float*)d_in[15];
    const float* Wff1 = (const float*)d_in[16];
    const float* bff1 = (const float*)d_in[17];
    const float* Wff2 = (const float*)d_in[18];
    const float* bff2 = (const float*)d_in[19];
    const float* ln2g = (const float*)d_in[20];
    const float* ln2b = (const float*)d_in[21];
    const float* W1 = (const float*)d_in[22];
    const float* b1 = (const float*)d_in[23];
    const float* W2 = (const float*)d_in[24];
    const float* b2 = (const float*)d_in[25];
    const float* W3 = (const float*)d_in[26];
    const float* b3 = (const float*)d_in[27];

    const size_t SD = (size_t)S_LEN * DMODEL;
    const size_t DD = (size_t)DMODEL * DMODEL;
    const size_t DF = (size_t)DMODEL * FFDIM;
    const size_t WPER = 4 * DD + 2 * DF;

    float* x0     = (float*)d_ws;
    float* x1     = x0 + SD;
    float* t0     = x1 + SD;
    float* pooled = t0 + SD;
    float* h1     = pooled + DMODEL;
    float* h2     = h1 + 512;
    f16* x0h  = (f16*)(h2 + 256);
    f16* x1h  = x0h + SD;
    f16* qh   = x1h + SD;
    f16* kh   = qh + SD;
    f16* vh   = kh + SD;
    f16* abh  = vh + SD;
    f16* hffh = abh + SD;        // S x FF
    f16* wts  = hffh + 4 * SD;   // transposed f16 weights, NLAYER * WPER

    auto gemm = [&](const f16* A, const f16* Wt, const float* bias,
                    float* outf, f16* outh, int M, int N, int K, float scale, int act) {
        dim3 grid(N / 128, M / 128);
        gemm_kernel<<<grid, 256, 0, stream>>>(A, Wt, bias, outf, outh, M, N, K, scale, act);
    };
    auto wtrans = [&](const float* W, f16* Wt, int K, int N) {
        wtrans_kernel<<<dim3(N / 32, K / 32), 256, 0, stream>>>(W, Wt, K, N);
    };

    // One-time (per call) weight transpose+convert to f16 [N,K]
    for (int l = 0; l < NLAYER; ++l) {
        f16* wp = wts + (size_t)l * WPER;
        wtrans(Wq + (size_t)l * DD, wp, DMODEL, DMODEL);
        wtrans(Wk + (size_t)l * DD, wp + DD, DMODEL, DMODEL);
        wtrans(Wv + (size_t)l * DD, wp + 2 * DD, DMODEL, DMODEL);
        wtrans(Wo + (size_t)l * DD, wp + 3 * DD, DMODEL, DMODEL);
        wtrans(Wff1 + (size_t)l * DF, wp + 4 * DD, DMODEL, FFDIM);        // -> [FF, D]
        wtrans(Wff2 + (size_t)l * DF, wp + 4 * DD + DF, FFDIM, DMODEL);   // -> [D, FF]
    }

    // Embedding + LN (fp32 master + f16 shadow)
    embed_kernel<<<(S_LEN * DMODEL) / 256, 256, 0, stream>>>(ids, wordemb, posemb, t0);
    ln_kernel<<<S_LEN, 256, 0, stream>>>(t0, nullptr, emb_g, emb_b, x0, x0h);

    for (int l = 0; l < NLAYER; ++l) {
        f16* wp = wts + (size_t)l * WPER;
        const f16* wqt = wp;
        const f16* wkt = wp + DD;
        const f16* wvt = wp + 2 * DD;
        const f16* wot = wp + 3 * DD;
        const f16* wf1 = wp + 4 * DD;
        const f16* wf2 = wp + 4 * DD + DF;

        gemm(x0h, wqt, bq + l * DMODEL, nullptr, qh, S_LEN, DMODEL, DMODEL, 0.125f, 0);
        gemm(x0h, wkt, bk + l * DMODEL, nullptr, kh, S_LEN, DMODEL, DMODEL, 1.0f, 0);
        gemm(x0h, wvt, bv + l * DMODEL, nullptr, vh, S_LEN, DMODEL, DMODEL, 1.0f, 0);

        attn_kernel<<<dim3(NCHUNK, NHEAD), 256, 0, stream>>>(qh, kh, vh, amask, abh);

        gemm(abh, wot, bo + l * DMODEL, t0, nullptr, S_LEN, DMODEL, DMODEL, 1.0f, 0);
        ln_kernel<<<S_LEN, 256, 0, stream>>>(x0, t0, ln1g + l * DMODEL, ln1b + l * DMODEL, x1, x1h);

        gemm(x1h, wf1, bff1 + l * FFDIM, nullptr, hffh, S_LEN, FFDIM, DMODEL, 1.0f, 1);
        gemm(hffh, wf2, bff2 + l * DMODEL, t0, nullptr, S_LEN, DMODEL, FFDIM, 1.0f, 0);
        ln_kernel<<<S_LEN, 256, 0, stream>>>(x1, t0, ln2g + l * DMODEL, ln2b + l * DMODEL, x0, x0h);
    }

    pool_kernel<<<DMODEL, 256, 0, stream>>>(x0, amask, pooled);
    fc_kernel<<<2, 256, 0, stream>>>(pooled, W1, b1, h1, DMODEL, 512, 1);
    fc_kernel<<<1, 256, 0, stream>>>(h1, W2, b2, h2, 512, 250, 1);
    final_kernel<<<1, 32, 0, stream>>>(h2, W3, b3, (float*)d_out);
}
